// MixLinear_GEMM_36859409334784
// MI455X (gfx1250) — 8-bit matrix operand order repaired and checked by reading, compiled, not run on hardware
//
#include <hip/hip_runtime.h>
#include <hip/hip_bf16.h>
#include <stdint.h>

#define SIGMA 20.0f
#define EPS_Q 1e-8f
#define MAX_OUTLIERS 256

#define BM 128
#define BN 128
#define BK 64

typedef __attribute__((ext_vector_type(8))) int v8i;
typedef int v4i_gv __attribute__((vector_size(16)));

#if __has_builtin(__builtin_amdgcn_global_load_async_to_lds_b128)
#define HAVE_ASYNC_LDS 1
#else
#define HAVE_ASYNC_LDS 0
#endif

#define AS_G(p) ((__attribute__((address_space(1))) v4i_gv*)(p))
#define AS_L(p) ((__attribute__((address_space(3))) v4i_gv*)(p))

__device__ __forceinline__ void wait_async_lds() {
#if HAVE_ASYNC_LDS
#if __has_builtin(__builtin_amdgcn_s_wait_asynccnt)
    __builtin_amdgcn_s_wait_asynccnt(0);
#else
    asm volatile("s_wait_asynccnt 0" ::: "memory");
#endif
#endif
}

__global__ void init_ws_kernel(unsigned int* __restrict__ mask, int* __restrict__ cnt,
                               int maskWords) {
    int t = blockIdx.x * blockDim.x + threadIdx.x;
    if (t < maskWords) mask[t] = 0u;
    if (t == 0) *cnt = 0;
}

__global__ void find_outliers_kernel(const float* __restrict__ x,
                                     unsigned int* __restrict__ mask, int M, int K) {
    int k = blockIdx.x * blockDim.x + threadIdx.x;
    if (k >= K) return;
    bool any = false;
    for (int m = 0; m < M; ++m)
        any |= (fabsf(x[(size_t)m * K + k]) > SIGMA);
    if (any) atomicOr(&mask[k >> 5], 1u << (k & 31));
}

__global__ void compact_outliers_kernel(const unsigned int* __restrict__ mask,
                                        int* __restrict__ idx, int* __restrict__ cnt, int K) {
    if (blockIdx.x == 0 && threadIdx.x == 0) {
        int c = 0;
        for (int k = 0; k < K; ++k)
            if ((mask[k >> 5] >> (k & 31)) & 1u) {
                if (c < MAX_OUTLIERS) idx[c] = k;
                ++c;
            }
        *cnt = (c < MAX_OUTLIERS) ? c : MAX_OUTLIERS;
    }
}

__global__ __launch_bounds__(256)
void quantize_weight_kernel(const float* __restrict__ w,
                            const unsigned int* __restrict__ mask,
                            float* __restrict__ q_scale, int8_t* __restrict__ qw, int K) {
    const int n = blockIdx.x;
    const float* row = w + (size_t)n * K;
    __shared__ float red[256];
    float mx = 0.f;
    for (int k = threadIdx.x; k < K; k += 256) {
        bool keep = !((mask[k >> 5] >> (k & 31)) & 1u);
        mx = fmaxf(mx, keep ? fabsf(row[k]) : 0.f);
    }
    red[threadIdx.x] = mx;
    __syncthreads();
    for (int s = 128; s > 0; s >>= 1) {
        if (threadIdx.x < s) red[threadIdx.x] = fmaxf(red[threadIdx.x], red[threadIdx.x + s]);
        __syncthreads();
    }
    const float scale = fmaxf(red[0] / 127.f, EPS_Q);
    if (threadIdx.x == 0) q_scale[n] = scale;
    const float inv = 1.f / scale;
    for (int k = threadIdx.x; k < K; k += 256) {
        bool keep = !((mask[k >> 5] >> (k & 31)) & 1u);
        float v = keep ? row[k] * inv : 0.f;
        v = fminf(fmaxf(v, -128.f), 127.f);
        qw[(size_t)n * K + k] = (int8_t)__float2int_rn(v);
    }
}

__global__ __launch_bounds__(256)
void quantize_act_kernel(const float* __restrict__ x,
                         float* __restrict__ row_scale, int8_t* __restrict__ qx, int K) {
    const int m = blockIdx.x;
    const float* row = x + (size_t)m * K;
    __shared__ float red[256];
    float mx = 0.f;
    for (int k = threadIdx.x; k < K; k += 256)
        mx = fmaxf(mx, fabsf(row[k]));
    red[threadIdx.x] = mx;
    __syncthreads();
    for (int s = 128; s > 0; s >>= 1) {
        if (threadIdx.x < s) red[threadIdx.x] = fmaxf(red[threadIdx.x], red[threadIdx.x + s]);
        __syncthreads();
    }
    const float scale = fmaxf(red[0] / 127.f, EPS_Q);
    if (threadIdx.x == 0) row_scale[m] = scale;
    const float inv = 1.f / scale;
    for (int k = threadIdx.x; k < K; k += 256) {
        float v = fminf(fmaxf(row[k] * inv, -128.f), 127.f);
        qx[(size_t)m * K + k] = (int8_t)__float2int_rn(v);
    }
}

__global__ __launch_bounds__(256)
void gemm_i8_wmma_kernel(const int8_t* __restrict__ qx, const int8_t* __restrict__ qw,
                         const float* __restrict__ row_scale, const float* __restrict__ q_scale,
                         const float* __restrict__ xf, const float* __restrict__ w,
                         const float* __restrict__ bias,
                         const int* __restrict__ out_idx, const int* __restrict__ out_cnt,
                         float* __restrict__ y, int M, int N, int K) {
    __shared__ int8_t As[2][BM * BK];
    __shared__ int8_t Bs[2][BN * BK];

    const int tid  = threadIdx.x;
    const int wave = tid >> 5;
    const int lane = tid & 31;
    const int waveMoff = (wave >> 1) << 5;
    const int waveNoff = (wave & 1) << 6;
    const int rowBase = blockIdx.y * BM;
    const int colBase = blockIdx.x * BN;

    const int ldr = tid >> 1;
    const int ldc = (tid & 1) << 5;
    const int8_t* gA = qx + (size_t)(rowBase + ldr) * K + ldc;
    const int8_t* gB = qw + (size_t)(colBase + ldr) * K + ldc;

    const int lane15   = lane & 15;
    const int laneHalf = lane >> 4;

    v8i acc[2][4];
#pragma unroll
    for (int mi = 0; mi < 2; ++mi)
#pragma unroll
        for (int ni = 0; ni < 4; ++ni)
            acc[mi][ni] = (v8i){0, 0, 0, 0, 0, 0, 0, 0};

    auto stage = [&](int buf, int kk) {
        int8_t* lA = &As[buf][ldr * BK + ldc];
        int8_t* lB = &Bs[buf][ldr * BK + ldc];
#if HAVE_ASYNC_LDS
#pragma unroll
        for (int j = 0; j < 2; ++j) {
            __builtin_amdgcn_global_load_async_to_lds_b128(AS_G(gA + kk + 16 * j),
                                                           AS_L(lA + 16 * j), 0, 0);
            __builtin_amdgcn_global_load_async_to_lds_b128(AS_G(gB + kk + 16 * j),
                                                           AS_L(lB + 16 * j), 0, 0);
        }
#else
        *(int4*)(lA +  0) = *(const int4*)(gA + kk +  0);
        *(int4*)(lA + 16) = *(const int4*)(gA + kk + 16);
        *(int4*)(lB +  0) = *(const int4*)(gB + kk +  0);
        *(int4*)(lB + 16) = *(const int4*)(gB + kk + 16);
        __builtin_prefetch(gA + kk + BK, 0, 1);
        __builtin_prefetch(gB + kk + BK, 0, 1);
#endif
    };

    stage(0, 0);
    wait_async_lds();
    __syncthreads();

    int buf = 0;
    for (int kk = 0; kk < K; kk += BK, buf ^= 1) {
        if (kk + BK < K) stage(buf ^ 1, kk + BK);

        const int8_t* lA = As[buf];
        const int8_t* lB = Bs[buf];

        v8i afrag[2];
#pragma unroll
        for (int mi = 0; mi < 2; ++mi) {
            const int aRow = waveMoff + (mi << 4) + lane15;
#pragma unroll
            for (int j = 0; j < 8; ++j) {
                const int kb = ((j >> 1) << 4) + ((j & 1) << 2) + (laneHalf << 3);
                afrag[mi][j] = *(const int*)(lA + aRow * BK + kb);
            }
        }
#pragma unroll
        for (int ni = 0; ni < 4; ++ni) {
            const int bRow = waveNoff + (ni << 4) + lane15;
            v8i bfrag;
#pragma unroll
            for (int j = 0; j < 8; ++j) {
                const int kb = ((j >> 1) << 4) + ((j & 1) << 2) + (laneHalf << 3);
                bfrag[j] = *(const int*)(lB + bRow * BK + kb);
            }
            acc[0][ni] = __builtin_amdgcn_wmma_i32_16x16x64_iu8(true, afrag[0], true, bfrag,
                                                                acc[0][ni], false, false);
            acc[1][ni] = __builtin_amdgcn_wmma_i32_16x16x64_iu8(true, afrag[1], true, bfrag,
                                                                acc[1][ni], false, false);
        }

        wait_async_lds();
        __syncthreads();
    }

    const int cnt = *out_cnt;
#pragma unroll
    for (int mi = 0; mi < 2; ++mi) {
#pragma unroll
        for (int r = 0; r < 8; ++r) {
            const int   gm = rowBase + waveMoff + (mi << 4) + r + (laneHalf << 3);
            const float rs = row_scale[gm];
#pragma unroll
            for (int ni = 0; ni < 4; ++ni) {
                const int gn = colBase + waveNoff + (ni << 4) + lane15;
                float f = (float)acc[mi][ni][r] * rs * q_scale[gn];
                for (int i = 0; i < cnt; ++i) {
                    const int k = out_idx[i];
                    f += xf[(size_t)gm * K + k] * w[(size_t)gn * K + k];
                }
                y[(size_t)gm * N + gn] = f + bias[gn];
            }
        }
    }
}

extern "C" void kernel_launch(void* const* d_in, const int* in_sizes, int n_in,
                              void* d_out, int out_size, void* d_ws, size_t ws_size,
                              hipStream_t stream) {
    const float* x    = (const float*)d_in[0];
    const float* w    = (const float*)d_in[1];
    const float* bias = (const float*)d_in[2];
    float*       y    = (float*)d_out;

    const int N = in_sizes[2];
    const int K = in_sizes[1] / N;
    const int M = in_sizes[0] / K;

    char*         ws        = (char*)d_ws;
    unsigned int* mask      = (unsigned int*)ws;
    const int     maskWords = (K + 31) / 32;
    int*          cnt       = (int*)(ws + 1024);
    int*          idx       = (int*)(ws + 2048);
    float*        row_scale = (float*)(ws + 4096);
    float*        q_scale   = (float*)(ws + 4096 + (size_t)M * 4);
    int8_t*       qx        = (int8_t*)(ws + 4096 + (size_t)(M + N) * 4);
    int8_t*       qw        = qx + (size_t)M * K;

    hipLaunchKernelGGL(init_ws_kernel, dim3((maskWords + 255) / 256), dim3(256), 0, stream,
                       mask, cnt, maskWords);
    hipLaunchKernelGGL(find_outliers_kernel, dim3((K + 255) / 256), dim3(256), 0, stream,
                       x, mask, M, K);
    hipLaunchKernelGGL(compact_outliers_kernel, dim3(1), dim3(1), 0, stream, mask, idx, cnt, K);
    hipLaunchKernelGGL(quantize_weight_kernel, dim3(N), dim3(256), 0, stream,
                       w, mask, q_scale, qw, K);
    hipLaunchKernelGGL(quantize_act_kernel, dim3(M), dim3(256), 0, stream,
                       x, row_scale, qx, K);
    hipLaunchKernelGGL(gemm_i8_wmma_kernel, dim3(N / BN, M / BM), dim3(256), 0, stream,
                       qx, qw, row_scale, q_scale, x, w, bias, idx, cnt, y, M, N, K);
}
